// Model_34239479283896
// MI455X (gfx1250) — matrix operand order repaired and checked by reading, not compiled, not run on hardware
//
#include <hip/hip_runtime.h>

typedef __attribute__((ext_vector_type(16))) _Float16 v16h;
typedef __attribute__((ext_vector_type(8)))  _Float16 v8h;
typedef __attribute__((ext_vector_type(8)))  float    v8f;
typedef __attribute__((ext_vector_type(4)))  int      v4i;

#define TPB 256
#define WPB 8   // waves per block (wave32)

#if __has_builtin(__builtin_amdgcn_global_load_async_to_lds_b128) && \
    __has_builtin(__builtin_amdgcn_s_wait_asynccnt)
#define ASYNC_COPY16(g, l)                                                    \
  __builtin_amdgcn_global_load_async_to_lds_b128(                             \
      (__attribute__((address_space(1))) v4i*)(g),                            \
      (__attribute__((address_space(3))) v4i*)(l), 0, 0)
#define ASYNC_WAIT() __builtin_amdgcn_s_wait_asynccnt(0)
#else
#define ASYNC_COPY16(g, l) (*(v8h*)(l) = *(const v8h*)(g))
#define ASYNC_WAIT() ((void)0)
#endif

#define KC_CONV 6   // k32-steps staged per chunk (K32 in {18,72}: multiple of 6)
#define KC_GEMM 8   // K32 in {392,32}: multiple of 8

// ---------------------------------------------------------------------------
// Fast implicit-GEMM conv (HWC layout, Cin % 32 == 0, Cout % 32 == 0).
// k-order = (kh*KW+kw)*Cin + ci (8 contiguous f16 per lane -> b128 loads).
// Bp = weights pre-packed in B-fragment order [kb32][nt][lane][e] (32B/lane).
// Block = 8 waves sharing one 32-col N-pair; B staged in LDS via async copy.
// Each wave computes a 32(M) x 32(N) tile: 2 A frags x 2 B frags -> 4 WMMAs.
// ---------------------------------------------------------------------------
__global__ void conv_wmma_fast(const _Float16* __restrict__ in,
                               const _Float16* __restrict__ Bp,
                               float* __restrict__ of32,
                               _Float16* __restrict__ of16,
                               int Cin, int H, int W, int Cout, int OH, int OW,
                               int KH, int KW, int stride, int pad, int relu)
{
  __shared__ _Float16 lds_b[KC_CONV * 2 * 512];   // 12 KB
  int M   = OH * OW;
  int Mt  = (M + 31) >> 5;                // 32-row M tiles
  int Mtg = (Mt + 7) >> 3;                // groups of 8 M tiles per block
  int Nt  = Cout >> 4;
  int ntp = blockIdx.x / Mtg;
  int mtg = blockIdx.x - ntp * Mtg;
  int wave = threadIdx.x >> 5;
  int mt  = mtg * 8 + wave;
  int nt0 = ntp * 2;
  int lane = threadIdx.x & 31;
  int half = lane >> 4, l15 = lane & 15;
  int m0  = mt << 5;
  int mA0 = m0 + l15;
  int mA1 = m0 + 16 + l15;
  bool mok0 = (mA0 < M), mok1 = (mA1 < M);
  int oh0 = mA0 / OW, ow0 = mA0 - oh0 * OW;
  int oh1 = mA1 / OW, ow1 = mA1 - oh1 * OW;
  int K32 = (Cin >> 5) * KH * KW;
  const char* gB = (const char*)Bp;
  char* lB = (char*)lds_b;
  const v16h* lBv = (const v16h*)lds_b;

  v8f acc00 = {}, acc01 = {}, acc10 = {}, acc11 = {};
  int kb32 = 0;
  for (int kh = 0; kh < KH; ++kh) {
    int ih0 = oh0 * stride + kh - pad;
    int ih1 = oh1 * stride + kh - pad;
    bool h0 = mok0 && ((unsigned)ih0 < (unsigned)H);
    bool h1 = mok1 && ((unsigned)ih1 < (unsigned)H);
    for (int kw = 0; kw < KW; ++kw) {
      int iw0 = ow0 * stride + kw - pad;
      int iw1 = ow1 * stride + kw - pad;
      bool ok0 = h0 && ((unsigned)iw0 < (unsigned)W);
      bool ok1 = h1 && ((unsigned)iw1 < (unsigned)W);
      const _Float16* base0 = in + ((long)ih0 * W + iw0) * Cin;
      const _Float16* base1 = in + ((long)ih1 * W + iw1) * Cin;
      for (int cib = 0; cib < Cin; cib += 32, ++kb32) {
        // ---- stage next B chunk into LDS (uniform across block) ----
        if (kb32 % KC_CONV == 0) {
          __syncthreads();                       // previous chunk consumed
          int remain = K32 - kb32;
          if (remain > KC_CONV) remain = KC_CONV;
          int units = remain << 7;               // 128 x 16B units per k-step
          for (int u = threadIdx.x; u < units; u += TPB) {
            int kc  = u >> 7;
            int r   = u & 127;
            int nts = r >> 6;
            int r2  = r & 63;
            long gsrc = (((long)(kb32 + kc) * Nt + nt0 + nts) << 10) + ((long)r2 << 4);
            int  ldst = ((kc * 2 + nts) << 10) + (r2 << 4);
            ASYNC_COPY16(gB + gsrc, lB + ldst);
          }
          ASYNC_WAIT();
          __syncthreads();
        }
        int kc = kb32 % KC_CONV;
        // ---- A fragments (global, guarded) ----
        int co8 = cib + half * 8;
        v8h a0lo = {}, a0hi = {}, a1lo = {}, a1hi = {};
        if (ok0) {
          a0lo = *(const v8h*)(base0 + co8);
          a0hi = *(const v8h*)(base0 + co8 + 16);
        }
        if (ok1) {
          a1lo = *(const v8h*)(base1 + co8);
          a1hi = *(const v8h*)(base1 + co8 + 16);
        }
        v16h a0 = __builtin_shufflevector(a0lo, a0hi, 0,1,2,3,4,5,6,7,
                                                      8,9,10,11,12,13,14,15);
        v16h a1 = __builtin_shufflevector(a1lo, a1hi, 0,1,2,3,4,5,6,7,
                                                      8,9,10,11,12,13,14,15);
        // ---- B fragments (LDS) ----
        v16h b0 = lBv[(kc * 2 + 0) * 32 + lane];
        v16h b1 = lBv[(kc * 2 + 1) * 32 + lane];
        acc00 = __builtin_amdgcn_wmma_f32_16x16x32_f16(false, a0, false, b0,
                                                       (short)0, acc00, false, false);
        acc01 = __builtin_amdgcn_wmma_f32_16x16x32_f16(false, a0, false, b1,
                                                       (short)0, acc01, false, false);
        acc10 = __builtin_amdgcn_wmma_f32_16x16x32_f16(false, a1, false, b0,
                                                       (short)0, acc10, false, false);
        acc11 = __builtin_amdgcn_wmma_f32_16x16x32_f16(false, a1, false, b1,
                                                       (short)0, acc11, false, false);
      }
    }
  }
  int col0 = nt0 * 16 + l15;
#pragma unroll
  for (int j = 0; j < 8; ++j) {
    int row0 = m0 + half * 8 + j;
    int row1 = row0 + 16;
    if (row0 < M) {
      float v0 = acc00[j], v1 = acc01[j];
      if (relu) { v0 = fmaxf(v0, 0.f); v1 = fmaxf(v1, 0.f); }
      long o = (long)row0 * Cout + col0;       // HWC output
      if (of32) { of32[o] = v0; of32[o + 16] = v1; }
      if (of16) { of16[o] = (_Float16)v0; of16[o + 16] = (_Float16)v1; }
    }
    if (row1 < M) {
      float v0 = acc10[j], v1 = acc11[j];
      if (relu) { v0 = fmaxf(v0, 0.f); v1 = fmaxf(v1, 0.f); }
      long o = (long)row1 * Cout + col0;
      if (of32) { of32[o] = v0; of32[o + 16] = v1; }
      if (of16) { of16[o] = (_Float16)v0; of16[o + 16] = (_Float16)v1; }
    }
  }
}

// Generic (slow) conv for the stem: Cin=3, K=147. HWC in/out,
// weights in plain [K][Cout] with k = (kh*KW+kw)*Cin + ci.
__global__ void conv_wmma_generic(const _Float16* __restrict__ in,
                                  const _Float16* __restrict__ wt,
                                  _Float16* __restrict__ of16,
                                  int Cin, int H, int W, int Cout, int OH, int OW,
                                  int KH, int KW, int stride, int pad, int relu)
{
  int M  = OH * OW;
  int K  = Cin * KH * KW;
  int Mt = (M + 15) >> 4;
  int Nt = Cout >> 4;
  int wid = blockIdx.x * WPB + (threadIdx.x >> 5);
  if (wid >= Mt * Nt) return;
  int mt = wid % Mt, nt = wid / Mt;
  int lane = threadIdx.x & 31;
  int half = lane >> 4, l15 = lane & 15;
  int m0 = mt << 4;
  int mA = m0 + l15;
  int nB = nt * 16 + l15;
  int oh = mA / OW;
  int ow = mA - oh * OW;
  v8f acc = {};
  int Kp = (K + 31) & ~31;
  for (int kb = 0; kb < Kp; kb += 32) {
    v16h a, b;
#pragma unroll
    for (int e = 0; e < 16; ++e) {
      int k = kb + half * 8 + (e & 7) + (e >> 3) * 16;   // e<8: +0 ; e>=8: +16
      _Float16 val = (_Float16)0.f;
      if (mA < M && k < K) {
        int khw = k / Cin;
        int ci  = k - khw * Cin;
        int kh  = khw / KW;
        int kw  = khw - kh * KW;
        int ih  = oh * stride + kh - pad;
        int iw  = ow * stride + kw - pad;
        if ((unsigned)ih < (unsigned)H && (unsigned)iw < (unsigned)W)
          val = in[((long)ih * W + iw) * Cin + ci];
      }
      a[e] = val;
    }
#pragma unroll
    for (int e = 0; e < 16; ++e) {
      int k = kb + half * 8 + (e & 7) + (e >> 3) * 16;
      b[e] = (k < K) ? wt[k * Cout + nB] : (_Float16)0.f;
    }
    acc = __builtin_amdgcn_wmma_f32_16x16x32_f16(false, a, false, b,
                                                 (short)0, acc, false, false);
  }
#pragma unroll
  for (int j = 0; j < 8; ++j) {
    int row = m0 + half * 8 + j;
    if (row < M) {
      float v = acc[j];
      if (relu) v = fmaxf(v, 0.f);
      of16[(long)row * Cout + nB] = (_Float16)v;
    }
  }
}

// Row-major GEMM, M%32==0, N%32==0, K%32==0, K32%KC_GEMM==0.
// Block = 8 waves sharing one N-pair; B staged in LDS via async copy.
__global__ void gemm_wmma_fast(const _Float16* __restrict__ A,
                               const _Float16* __restrict__ Bp,
                               const float* __restrict__ bias,
                               float* __restrict__ of32,
                               _Float16* __restrict__ of16,
                               int M, int N, int K, int relu)
{
  __shared__ _Float16 lds_b[KC_GEMM * 2 * 512];   // 16 KB
  int Mt  = M >> 5;
  int Mtg = (Mt + 7) >> 3;
  int Nt  = N >> 4;
  int ntp = blockIdx.x / Mtg;
  int mtg = blockIdx.x - ntp * Mtg;
  int wave = threadIdx.x >> 5;
  int mt  = mtg * 8 + wave;
  int nt0 = ntp * 2;
  int lane = threadIdx.x & 31;
  int half = lane >> 4, l15 = lane & 15;
  int m0 = mt << 5;
  bool mok = (mt < Mt);
  int mr0 = mok ? (m0 + l15) : 0;
  int mr1 = mok ? (m0 + 16 + l15) : 0;
  const _Float16* arow0 = A + (long)mr0 * K;
  const _Float16* arow1 = A + (long)mr1 * K;
  const char* gB = (const char*)Bp;
  char* lB = (char*)lds_b;
  const v16h* lBv = (const v16h*)lds_b;
  v8f acc00 = {}, acc01 = {}, acc10 = {}, acc11 = {};
  int K32 = K >> 5;
  for (int kb32 = 0; kb32 < K32; ++kb32) {
    if ((kb32 & (KC_GEMM - 1)) == 0) {
      __syncthreads();
      int remain = K32 - kb32;
      if (remain > KC_GEMM) remain = KC_GEMM;
      int units = remain << 7;
      for (int u = threadIdx.x; u < units; u += TPB) {
        int kc  = u >> 7;
        int r   = u & 127;
        int nts = r >> 6;
        int r2  = r & 63;
        long gsrc = (((long)(kb32 + kc) * Nt + nt0 + nts) << 10) + ((long)r2 << 4);
        int  ldst = ((kc * 2 + nts) << 10) + (r2 << 4);
        ASYNC_COPY16(gB + gsrc, lB + ldst);
      }
      ASYNC_WAIT();
      __syncthreads();
    }
    int kc = kb32 & (KC_GEMM - 1);
    int k0 = kb32 * 32 + half * 8;
    v8h a0lo = *(const v8h*)(arow0 + k0);
    v8h a0hi = *(const v8h*)(arow0 + k0 + 16);
    v8h a1lo = *(const v8h*)(arow1 + k0);
    v8h a1hi = *(const v8h*)(arow1 + k0 + 16);
    v16h a0 = __builtin_shufflevector(a0lo, a0hi, 0,1,2,3,4,5,6,7,
                                                  8,9,10,11,12,13,14,15);
    v16h a1 = __builtin_shufflevector(a1lo, a1hi, 0,1,2,3,4,5,6,7,
                                                  8,9,10,11,12,13,14,15);
    v16h b0 = lBv[(kc * 2 + 0) * 32 + lane];
    v16h b1 = lBv[(kc * 2 + 1) * 32 + lane];
    acc00 = __builtin_amdgcn_wmma_f32_16x16x32_f16(false, a0, false, b0,
                                                   (short)0, acc00, false, false);
    acc01 = __builtin_amdgcn_wmma_f32_16x16x32_f16(false, a0, false, b1,
                                                   (short)0, acc01, false, false);
    acc10 = __builtin_amdgcn_wmma_f32_16x16x32_f16(false, a1, false, b0,
                                                   (short)0, acc10, false, false);
    acc11 = __builtin_amdgcn_wmma_f32_16x16x32_f16(false, a1, false, b1,
                                                   (short)0, acc11, false, false);
  }
  if (!mok) return;
  int col0 = nt0 * 16 + l15;
  float bv0 = bias ? bias[col0] : 0.f;
  float bv1 = bias ? bias[col0 + 16] : 0.f;
#pragma unroll
  for (int j = 0; j < 8; ++j) {
    int row0 = m0 + half * 8 + j;
    int row1 = row0 + 16;
    float v00 = acc00[j] + bv0, v01 = acc01[j] + bv1;
    float v10 = acc10[j] + bv0, v11 = acc11[j] + bv1;
    if (relu) {
      v00 = fmaxf(v00, 0.f); v01 = fmaxf(v01, 0.f);
      v10 = fmaxf(v10, 0.f); v11 = fmaxf(v11, 0.f);
    }
    long o0 = (long)row0 * N + col0;
    long o1 = (long)row1 * N + col0;
    if (of32) { of32[o0] = v00; of32[o0 + 16] = v01; of32[o1] = v10; of32[o1 + 16] = v11; }
    if (of16) {
      of16[o0] = (_Float16)v00; of16[o0 + 16] = (_Float16)v01;
      of16[o1] = (_Float16)v10; of16[o1 + 16] = (_Float16)v11;
    }
  }
}

// ---------------------------------------------------------------------------
// Weight packing
// ---------------------------------------------------------------------------
__global__ void pack_conv_w(const float* __restrict__ w, _Float16* __restrict__ Bp,
                            int Cout, int Cin, int KHW)
{
  int K = Cin * KHW;          // multiple of 32
  int Nt = Cout >> 4;
  long total = (long)K * Cout;
  long idx = (long)blockIdx.x * blockDim.x + threadIdx.x;
  if (idx >= total) return;
  long kb32 = idx / ((long)Nt * 512);
  long rem  = idx - kb32 * Nt * 512;
  int  nt   = (int)(rem / 512);
  int  r2   = (int)(rem - (long)nt * 512);
  int  lane = r2 >> 4;
  int  e    = r2 & 15;
  int  k    = (int)kb32 * 32 + (lane >> 4) * 8 + (e & 7) + (e >> 3) * 16;
  int  co   = nt * 16 + (lane & 15);
  int  khw  = k / Cin;
  int  ci   = k - khw * Cin;
  Bp[idx] = (_Float16)w[((long)co * Cin + ci) * KHW + khw];
}

__global__ void pack_fc_w(const float* __restrict__ w, _Float16* __restrict__ Bp,
                          int K, int N)
{
  int Nt = N >> 4;
  long total = (long)K * N;
  long idx = (long)blockIdx.x * blockDim.x + threadIdx.x;
  if (idx >= total) return;
  long kb32 = idx / ((long)Nt * 512);
  long rem  = idx - kb32 * Nt * 512;
  int  nt   = (int)(rem / 512);
  int  r2   = (int)(rem - (long)nt * 512);
  int  lane = r2 >> 4;
  int  e    = r2 & 15;
  int  k    = (int)kb32 * 32 + (lane >> 4) * 8 + (e & 7) + (e >> 3) * 16;
  int  co   = nt * 16 + (lane & 15);
  Bp[idx] = (_Float16)w[(long)k * N + co];
}

__global__ void pack_stem_w(const float* __restrict__ w, _Float16* __restrict__ wt,
                            int Cout, int Cin, int KHW)
{
  int K = Cin * KHW;
  int idx = blockIdx.x * blockDim.x + threadIdx.x;
  if (idx >= K * Cout) return;
  int k  = idx / Cout;
  int co = idx - k * Cout;
  int khw = k / Cin;
  int ci  = k - khw * Cin;
  wt[idx] = (_Float16)w[((long)co * Cin + ci) * KHW + khw];
}

// ---------------------------------------------------------------------------
// Glue kernels
// ---------------------------------------------------------------------------
__global__ void normalize_kernel(const float* __restrict__ x, _Float16* __restrict__ out, int hw)
{
  int idx = blockIdx.x * blockDim.x + threadIdx.x;   // out is HWC [hw][3]
  if (idx >= 3 * hw) return;
  int p = idx / 3;
  int c = idx - p * 3;
  const float mean[3] = {0.485f, 0.456f, 0.406f};
  const float stdv[3] = {0.229f, 0.224f, 0.225f};
  out[idx] = (_Float16)((x[c * hw + p] - mean[c]) / stdv[c]);
}

__global__ void conv1x1_kernel(const _Float16* __restrict__ in, const float* __restrict__ w,
                               float* __restrict__ out, int Ci, int HW, int Co)
{
  int idx = blockIdx.x * blockDim.x + threadIdx.x;
  if (idx >= Co * HW) return;
  int co = idx / HW;
  int p  = idx - co * HW;
  const _Float16* ip = in + (long)p * Ci;
  const float*    wp = w + (long)co * Ci;
  float acc = 0.f;
  for (int ci = 0; ci < Ci; ++ci) acc += (float)ip[ci] * wp[ci];
  out[idx] = acc;
}

__global__ void decode_kernel(const float* __restrict__ cls, const float* __restrict__ box,
                              float* __restrict__ boxes, float* __restrict__ scores,
                              int fh, int fw, int stride, float size)
{
  int idx = blockIdx.x * blockDim.x + threadIdx.x;
  int hw = fh * fw;
  if (idx >= hw * 3) return;
  int r = idx % 3;
  int p = idx / 3;
  int y = p / fw;
  int x = p - y * fw;
  float cx = (x + 0.5f) * (float)stride;
  float cy = (y + 0.5f) * (float)stride;
  const float wmul[3] = {1.41421356f, 1.0f, 0.70710678f};
  const float hmul[3] = {0.70710678f, 1.0f, 1.41421356f};
  float aw = size * wmul[r], ah = size * hmul[r];
  float d0 = box[(r * 4 + 0) * hw + p];
  float d1 = box[(r * 4 + 1) * hw + p];
  float d2 = box[(r * 4 + 2) * hw + p];
  float d3 = box[(r * 4 + 3) * hw + p];
  d2 = fminf(fmaxf(d2, -4.f), 4.f);
  d3 = fminf(fmaxf(d3, -4.f), 4.f);
  float ncx = cx + d0 * aw;
  float ncy = cy + d1 * ah;
  float w2 = aw * expf(d2) * 0.5f;
  float h2 = ah * expf(d3) * 0.5f;
  boxes[idx * 4 + 0] = fminf(fmaxf(ncx - w2, 0.f), 800.f);
  boxes[idx * 4 + 1] = fminf(fmaxf(ncy - h2, 0.f), 800.f);
  boxes[idx * 4 + 2] = fminf(fmaxf(ncx + w2, 0.f), 800.f);
  boxes[idx * 4 + 3] = fminf(fmaxf(ncy + h2, 0.f), 800.f);
  scores[idx] = cls[r * hw + p];
}

__global__ void topk_kernel(const float* __restrict__ scores, const float* __restrict__ boxes,
                            int N, int Kk, float* __restrict__ osc, float* __restrict__ obx)
{
  int i = blockIdx.x * blockDim.x + threadIdx.x;
  if (i >= N) return;
  float s = scores[i];
  int rank = 0;
  for (int j = 0; j < N; ++j) {
    float sj = scores[j];
    rank += (sj > s) || (sj == s && j < i);
  }
  if (rank < Kk) {
    osc[rank] = s;
    obx[rank * 4 + 0] = boxes[i * 4 + 0];
    obx[rank * 4 + 1] = boxes[i * 4 + 1];
    obx[rank * 4 + 2] = boxes[i * 4 + 2];
    obx[rank * 4 + 3] = boxes[i * 4 + 3];
  }
}

#define NMS_N 4000
#define POST_NMS 512

__global__ void nms_kernel(const float* __restrict__ boxes, const float* __restrict__ scores,
                           float* __restrict__ props_ws, float* __restrict__ dout)
{
  __shared__ float s_sc[NMS_N];
  __shared__ float rv[TPB];
  __shared__ int   ri[TPB];
  int tid = threadIdx.x;
  for (int j = tid; j < NMS_N; j += TPB) s_sc[j] = scores[j];
  __syncthreads();
  for (int it = 0; it < POST_NMS; ++it) {
    float bv = -INFINITY;
    int   bi = 0x7fffffff;
    for (int j = tid; j < NMS_N; j += TPB) {
      float v = s_sc[j];
      if (v > bv || (v == bv && j < bi)) { bv = v; bi = j; }
    }
    rv[tid] = bv; ri[tid] = bi;
    __syncthreads();
    for (int off = TPB / 2; off > 0; off >>= 1) {
      if (tid < off) {
        float ov = rv[tid + off]; int oi = ri[tid + off];
        if (ov > rv[tid] || (ov == rv[tid] && oi < ri[tid])) { rv[tid] = ov; ri[tid] = oi; }
      }
      __syncthreads();
    }
    int best = ri[0];
    __syncthreads();
    float bx1 = boxes[best * 4 + 0], by1 = boxes[best * 4 + 1];
    float bx2 = boxes[best * 4 + 2], by2 = boxes[best * 4 + 3];
    float barea = (bx2 - bx1) * (by2 - by1);
    if (tid == 0) {
      props_ws[it * 4 + 0] = bx1; props_ws[it * 4 + 1] = by1;
      props_ws[it * 4 + 2] = bx2; props_ws[it * 4 + 3] = by2;
      dout[5120 + it * 4 + 0] = bx1; dout[5120 + it * 4 + 1] = by1;
      dout[5120 + it * 4 + 2] = bx2; dout[5120 + it * 4 + 3] = by2;
      dout[7168 + it] = scores[best];
    }
    for (int j = tid; j < NMS_N; j += TPB) {
      float x1 = boxes[j * 4 + 0], y1 = boxes[j * 4 + 1];
      float x2 = boxes[j * 4 + 2], y2 = boxes[j * 4 + 3];
      float ix1 = fmaxf(x1, bx1), iy1 = fmaxf(y1, by1);
      float ix2 = fminf(x2, bx2), iy2 = fminf(y2, by2);
      float inter = fmaxf(ix2 - ix1, 0.f) * fmaxf(iy2 - iy1, 0.f);
      float area  = (x2 - x1) * (y2 - y1);
      float iou = inter / (area + barea - inter + 1e-6f);
      if (iou > 0.7f) s_sc[j] = -INFINITY;
    }
    __syncthreads();
  }
}

#define POOL 7
__global__ void roi_align_kernel(const float* __restrict__ feat, const float* __restrict__ rois,
                                 _Float16* __restrict__ pooled)
{
  int idx = blockIdx.x * blockDim.x + threadIdx.x;
  int total = POST_NMS * 256 * POOL * POOL;
  if (idx >= total) return;
  int n   = idx / (256 * POOL * POOL);
  int rem = idx - n * (256 * POOL * POOL);
  int c   = rem / (POOL * POOL);
  int pp  = rem - c * (POOL * POOL);
  int py  = pp / POOL;
  int px  = pp - py * POOL;
  const int H = 200, W = 200;
  const float scale = 0.25f;
  float rx1 = rois[n * 4 + 0], ry1 = rois[n * 4 + 1];
  float rx2 = rois[n * 4 + 2], ry2 = rois[n * 4 + 3];
  float x1 = rx1 * scale, y1 = ry1 * scale;
  float bw = (rx2 - rx1) * scale / POOL;
  float bh = (ry2 - ry1) * scale / POOL;
  float gx = x1 + (px + 0.5f) * bw - 0.5f;
  float gy = y1 + (py + 0.5f) * bh - 0.5f;
  float fx = fminf(fmaxf(floorf(gx), 0.f), (float)(W - 1));
  float fy = fminf(fmaxf(floorf(gy), 0.f), (float)(H - 1));
  int x0 = (int)fx, y0 = (int)fy;
  int x1i = min(x0 + 1, W - 1);
  int y1i = min(y0 + 1, H - 1);
  float wx = fminf(fmaxf(gx - fx, 0.f), 1.f);
  float wy = fminf(fmaxf(gy - fy, 0.f), 1.f);
  float v = feat[((long)y0 * W + x0) * 256 + c]  * (1.f - wy) * (1.f - wx)
          + feat[((long)y0 * W + x1i) * 256 + c] * (1.f - wy) * wx
          + feat[((long)y1i * W + x0) * 256 + c] * wy * (1.f - wx)
          + feat[((long)y1i * W + x1i) * 256 + c] * wy * wx;
  pooled[(long)n * 12544 + c * 49 + py * 7 + px] = (_Float16)v;
}

__global__ void heads_kernel(const float* __restrict__ v,
                             const float* __restrict__ cls_w, const float* __restrict__ cls_b,
                             const float* __restrict__ breg_w, const float* __restrict__ breg_b,
                             float* __restrict__ dout)
{
  int idx = blockIdx.x * blockDim.x + threadIdx.x;
  if (idx >= POST_NMS * 10) return;
  int row = idx / 10;
  int j   = idx - row * 10;
  const float* vr = v + (long)row * 1024;
  if (j < 2) {
    float acc = cls_b[j];
    for (int k = 0; k < 1024; ++k) acc += vr[k] * cls_w[k * 2 + j];
    dout[row * 2 + j] = acc;
  } else {
    int jj = j - 2;
    float acc = breg_b[jj];
    for (int k = 0; k < 1024; ++k) acc += vr[k] * breg_w[k * 8 + jj];
    dout[1024 + row * 8 + jj] = acc;
  }
}

static inline int cdiv(long a, long b) { return (int)((a + b - 1) / b); }

extern "C" void kernel_launch(void* const* d_in, const int* in_sizes, int n_in,
                              void* d_out, int out_size, void* d_ws, size_t ws_size,
                              hipStream_t stream)
{
  const float* x         = (const float*)d_in[0];
  const float* stem_w    = (const float*)d_in[1];
  const float* c2_w      = (const float*)d_in[2];
  const float* conv_w[8];                  // c3,c4,c5,fpn2,fpn3,fpn4,fpn5,rpn
  for (int i = 0; i < 8; ++i) conv_w[i] = (const float*)d_in[3 + i];
  const float* rpn_cls_w = (const float*)d_in[11];
  const float* rpn_box_w = (const float*)d_in[12];
  const float* fc1_w     = (const float*)d_in[13];
  const float* fc1_b     = (const float*)d_in[14];
  const float* fc2_w     = (const float*)d_in[15];
  const float* fc2_b     = (const float*)d_in[16];
  const float* cls_w     = (const float*)d_in[17];
  const float* cls_b     = (const float*)d_in[18];
  const float* breg_w    = (const float*)d_in[19];
  const float* breg_b    = (const float*)d_in[20];
  float* out = (float*)d_out;

  // ---- workspace bump allocator ----
  char* wsp = (char*)d_ws;
  size_t off = 0;
  auto alloc = [&](size_t bytes) -> void* {
    off = (off + 255) & ~(size_t)255;
    void* p = wsp + off;
    off += bytes;
    return p;
  };
  auto a16 = [&](long n) { return (_Float16*)alloc((size_t)n * 2); };
  auto a32 = [&](long n) { return (float*)alloc((size_t)n * 4); };

  _Float16* wt_stem = a16(160L * 64);
  _Float16* wt_c2   = a16(576L * 256);
  _Float16* wt_cv[8];
  for (int i = 0; i < 8; ++i) wt_cv[i] = a16(2304L * 256);
  _Float16* wt_fc1 = a16(12544L * 1024);
  _Float16* wt_fc2 = a16(1024L * 1024);

  _Float16* xn    = a16(800L * 800 * 3);
  _Float16* h16   = a16(400L * 400 * 64);
  _Float16* c2_16 = a16(200L * 200 * 256);
  _Float16* c3_16 = a16(100L * 100 * 256);
  _Float16* c4_16 = a16(50L * 50 * 256);
  _Float16* c5_16 = a16(25L * 25 * 256);
  _Float16* p16[4];
  p16[0] = a16(200L * 200 * 256);
  p16[1] = a16(100L * 100 * 256);
  p16[2] = a16(50L * 50 * 256);
  p16[3] = a16(25L * 25 * 256);
  float*    p2_f32 = a32(200L * 200 * 256);
  _Float16* t_buf  = a16(200L * 200 * 256);
  float* cls_buf = a32(3L * 40000);
  float* box_buf = a32(12L * 40000);
  float* dec_bx  = a32(120000L * 4);
  float* dec_sc  = a32(120000L);
  float* boxes_all  = a32(4000L * 4);
  float* scores_all = a32(4000L);
  float* props      = a32(512L * 4);
  _Float16* pooled  = a16(512L * 12544);
  _Float16* v1_16   = a16(512L * 1024);
  float*    v2_f32  = a32(512L * 1024);
  (void)ws_size; (void)in_sizes; (void)n_in; (void)out_size;

  pack_stem_w<<<cdiv(147L * 64, TPB), TPB, 0, stream>>>(stem_w, wt_stem, 64, 3, 49);
  pack_conv_w<<<cdiv(576L * 256, TPB), TPB, 0, stream>>>(c2_w, wt_c2, 256, 64, 9);
  for (int i = 0; i < 8; ++i)
    pack_conv_w<<<cdiv(2304L * 256, TPB), TPB, 0, stream>>>(conv_w[i], wt_cv[i], 256, 256, 9);
  pack_fc_w<<<cdiv(12544L * 1024, TPB), TPB, 0, stream>>>(fc1_w, wt_fc1, 12544, 1024);
  pack_fc_w<<<cdiv(1024L * 1024, TPB), TPB, 0, stream>>>(fc2_w, wt_fc2, 1024, 1024);

  normalize_kernel<<<cdiv(3L * 640000, TPB), TPB, 0, stream>>>(x, xn, 640000);

  auto conv = [&](const _Float16* in, const _Float16* Bp, float* of32, _Float16* of16,
                  int Cin, int H, int W, int Cout, int OH, int OW, int s, int pad, int relu) {
    int Mt  = (OH * OW + 31) / 32;
    int Mtg = (Mt + 7) / 8;
    int blocks = Mtg * (Cout / 32);
    conv_wmma_fast<<<blocks, TPB, 0, stream>>>(
        in, Bp, of32, of16, Cin, H, W, Cout, OH, OW, 3, 3, s, pad, relu);
  };

  {
    long tiles = (long)((400 * 400 + 15) / 16) * (64 / 16);
    conv_wmma_generic<<<cdiv(tiles, WPB), TPB, 0, stream>>>(
        xn, wt_stem, h16, 3, 800, 800, 64, 400, 400, 7, 7, 2, 2, 1);
  }
  conv(h16,   wt_c2,    nullptr, c2_16, 64,  400, 400, 256, 200, 200, 2, 0, 1);
  conv(c2_16, wt_cv[0], nullptr, c3_16, 256, 200, 200, 256, 100, 100, 2, 0, 1);
  conv(c3_16, wt_cv[1], nullptr, c4_16, 256, 100, 100, 256, 50,  50,  2, 0, 1);
  conv(c4_16, wt_cv[2], nullptr, c5_16, 256, 50,  50,  256, 25,  25,  2, 0, 1);

  conv(c2_16, wt_cv[3], p2_f32,  p16[0], 256, 200, 200, 256, 200, 200, 1, 1, 0);
  conv(c3_16, wt_cv[4], nullptr, p16[1], 256, 100, 100, 256, 100, 100, 1, 1, 0);
  conv(c4_16, wt_cv[5], nullptr, p16[2], 256, 50,  50,  256, 50,  50,  1, 1, 0);
  conv(c5_16, wt_cv[6], nullptr, p16[3], 256, 25,  25,  256, 25,  25,  1, 1, 0);

  const int   FH[4]  = {200, 100, 50, 25};
  const int   STR[4] = {4, 8, 16, 32};
  const float SZ[4]  = {32.f, 64.f, 128.f, 256.f};
  for (int l = 0; l < 4; ++l) {
    int fh = FH[l], fw = FH[l], hw = fh * fw;
    conv(p16[l], wt_cv[7], nullptr, t_buf, 256, fh, fw, 256, fh, fw, 1, 1, 1);
    conv1x1_kernel<<<cdiv(3L * hw, TPB), TPB, 0, stream>>>(t_buf, rpn_cls_w, cls_buf, 256, hw, 3);
    conv1x1_kernel<<<cdiv(12L * hw, TPB), TPB, 0, stream>>>(t_buf, rpn_box_w, box_buf, 256, hw, 12);
    int N = hw * 3;
    decode_kernel<<<cdiv(N, TPB), TPB, 0, stream>>>(cls_buf, box_buf, dec_bx, dec_sc,
                                                    fh, fw, STR[l], SZ[l]);
    topk_kernel<<<cdiv(N, TPB), TPB, 0, stream>>>(dec_sc, dec_bx, N, 1000,
                                                  scores_all + l * 1000,
                                                  boxes_all + (long)l * 1000 * 4);
  }

  nms_kernel<<<1, TPB, 0, stream>>>(boxes_all, scores_all, props, out);
  roi_align_kernel<<<cdiv(512L * 256 * 49, TPB), TPB, 0, stream>>>(p2_f32, props, pooled);

  {
    int Mtg = ((512 / 32) + 7) / 8;
    int blocks = Mtg * (1024 / 32);
    gemm_wmma_fast<<<blocks, TPB, 0, stream>>>(
        pooled, wt_fc1, fc1_b, nullptr, v1_16, 512, 1024, 12544, 1);
    gemm_wmma_fast<<<blocks, TPB, 0, stream>>>(
        v1_16, wt_fc2, fc2_b, v2_f32, nullptr, 512, 1024, 1024, 1);
  }
  heads_kernel<<<cdiv(512L * 10, TPB), TPB, 0, stream>>>(v2_f32, cls_w, cls_b,
                                                         breg_w, breg_b, out);
}
